// TemplatePairStackIteration_2199023255614
// MI455X (gfx1250) — matrix operand order repaired and checked by reading, not compiled, not run on hardware
//
// TemplatePairStackIteration for MI455X (gfx1250, wave32, WMMA bf16 16x16x32).
// All GEMM-shaped work -> v_wmma_f32_16x16x32_bf16 (f32 accum), elementwise in f32.
// Round 1: templated tile shapes/modes so accumulators stay in VGPRs (round-0
// disasm showed acc[] spilled to scratch because ntc/mode were runtime values).
#include <hip/hip_runtime.h>
#include <hip/hip_bf16.h>
#include <stdint.h>

#define NSEQ 192
#define R2   36864      // 192*192
typedef __bf16 bf16;
typedef __attribute__((ext_vector_type(16))) __bf16 bf16x16;
typedef __attribute__((ext_vector_type(8)))  float  f32x8;

__device__ __forceinline__ bf16 f2bf(float f) {
  uint32_t u = __float_as_uint(f);
  u += 0x7FFFu + ((u >> 16) & 1u);
  uint16_t h = (uint16_t)(u >> 16);
  bf16 r; __builtin_memcpy(&r, &h, 2);
  return r;
}

// A-fragment (16x32 bf16): lane<16 -> row=lane, k = 0..15; lane>=16 -> row=lane-16, k = 16..31.
// Caller passes p = rowptr + (lane<16 ? 0 : 16); we read p[0..15].
__device__ __forceinline__ bf16x16 ldA(const bf16* p) {
  bf16x16 f;
#pragma unroll
  for (int e = 0; e < 8; ++e) { f[e] = p[e]; f[e + 8] = p[8 + e]; }
  return f;
}
// B-fragment (32x16 bf16) of B = R^T with R row-major [16 rows, 32 k]:
// lane<16 -> col=lane, k=0..15 ; lane>=16 -> col=lane-16, k=16..31.
// Caller passes p = R_rowptr(col) + (lane<16 ? 0 : 16); 16 contiguous bf16.
__device__ __forceinline__ bf16x16 ldB16(const bf16* p) {
  bf16x16 f;
#pragma unroll
  for (int e = 0; e < 16; ++e) f[e] = p[e];
  return f;
}
__device__ __forceinline__ f32x8 wmma_bf16(bf16x16 a, bf16x16 b, f32x8 c) {
  return __builtin_amdgcn_wmma_f32_16x16x32_bf16(false, a, false, b, (short)0, c, false, false);
}
__device__ __forceinline__ f32x8 zero8() {
  f32x8 z;
#pragma unroll
  for (int e = 0; e < 8; ++e) z[e] = 0.f;
  return z;
}
__device__ __forceinline__ float sigmoidf_(float v) { return 1.f / (1.f + __expf(-v)); }

// ---------------- weight packing ----------------
// dst[((kb*ntc+nt)*32+lane)*16+e] = W[kb*32 + (lane<16?0:16) + e][nt*16 + (lane&15)]
// perm 1: logical col n=h*32+c maps to source col c*4+h (head-major q/k/v/g outputs)
// perm 2: logical row k=h*32+c maps to source row c*4+h (wo consuming head-major obuf)
__global__ void pack_kernel(const float* __restrict__ W, bf16* __restrict__ dst,
                            int Kreal, int Nreal, int ntc, int perm, int total) {
  int idx = blockIdx.x * 256 + threadIdx.x;
  if (idx >= total) return;
  int e = idx & 15;
  int lane = (idx >> 4) & 31;
  int tile = idx >> 9;
  int nt = tile % ntc, kb = tile / ntc;
  int k = kb * 32 + ((lane < 16) ? 0 : 16) + e;
  int n = nt * 16 + (lane & 15);
  float v = 0.f;
  if (k < Kreal && n < Nreal) {
    int wr = k, wc = n;
    if (perm == 1) wc = (n & 31) * 4 + (n >> 5);
    if (perm == 2) wr = (k & 31) * 4 + (k >> 5);
    v = W[(long)wr * Nreal + wc];
  }
  dst[idx] = f2bf(v);
}

// ---------------- layernorm (rows of 64) ----------------
__global__ __launch_bounds__(256)
void ln64_kernel(const float* __restrict__ x, const float* __restrict__ sc,
                 const float* __restrict__ bi, bf16* __restrict__ out) {
  int row  = blockIdx.x * 8 + (threadIdx.x >> 5);
  int lane = threadIdx.x & 31;
  float2 v = *(const float2*)(x + (long)row * 64 + lane * 2);
  float s = v.x + v.y, q = v.x * v.x + v.y * v.y;
#pragma unroll
  for (int xm = 1; xm <= 16; xm <<= 1) { s += __shfl_xor(s, xm, 32); q += __shfl_xor(q, xm, 32); }
  float mean = s * (1.f / 64.f);
  float inv  = rsqrtf(q * (1.f / 64.f) - mean * mean + 1e-5f);
  out[(long)row * 64 + lane * 2]     = f2bf((v.x - mean) * inv * sc[lane * 2]     + bi[lane * 2]);
  out[(long)row * 64 + lane * 2 + 1] = f2bf((v.y - mean) * inv * sc[lane * 2 + 1] + bi[lane * 2 + 1]);
}

// ---------------- generic WMMA GEMM, M=36864, K=32*KBLKS, N=16*NTC ----------------
constexpr int M_F32T4 = 0, M_BF16_SPLITH = 1, M_SIG_SPLITH = 2, M_SIG_PLAIN = 3,
              M_RELU128 = 4, M_RESID = 5, M_RESID_GATE = 6;

template <int KBLKS, int NTC, int MODE, int BIASMODE>
__global__ __launch_bounds__(256)
void gemm_kernel(const bf16* __restrict__ A, int arstride, long akstride,
                 const bf16* __restrict__ Bpk,
                 const float* __restrict__ bias,
                 float* __restrict__ outF, bf16* __restrict__ outB,
                 const float* __restrict__ gate) {
  int lane = threadIdx.x & 31;
  int rt = blockIdx.x * 8 + (threadIdx.x >> 5);  // row-tile (16 rows), 2304 total
  if (rt >= R2 / 16) return;
  int ml = lane & 15;
  int ka = (lane < 16) ? 0 : 16;
  f32x8 acc[NTC];
#pragma unroll
  for (int t = 0; t < NTC; ++t) acc[t] = zero8();
#pragma unroll
  for (int kb = 0; kb < KBLKS; ++kb) {
    const bf16* ap = A + (long)(rt * 16 + ml) * arstride + (long)kb * akstride + ka;
    bf16x16 af = ldA(ap);
#pragma unroll
    for (int nt = 0; nt < NTC; ++nt) {
      const bf16* bp = Bpk + (((long)kb * NTC + nt) * 32 + lane) * 16;
      acc[nt] = wmma_bf16(af, ldB16(bp), acc[nt]);
    }
  }
  int mrow = (lane < 16) ? 0 : 8;
#pragma unroll
  for (int nt = 0; nt < NTC; ++nt) {
#pragma unroll
    for (int e = 0; e < 8; ++e) {
      int r = rt * 16 + mrow + e;
      int n = nt * 16 + ml;
      float v = acc[nt][e];
      if (BIASMODE == 1) v += bias[n];
      else if (BIASMODE == 2) v += bias[(n & 31) * 4 + (n >> 5)];
      if (MODE == M_F32T4)            { if (n < 4) outF[(long)n * R2 + r] = v; }
      else if (MODE == M_BF16_SPLITH) outB[(long)(n >> 5) * R2 * 32 + (long)r * 32 + (n & 31)] = f2bf(v);
      else if (MODE == M_SIG_SPLITH)  outF[(long)(n >> 5) * R2 * 32 + (long)r * 32 + (n & 31)] = sigmoidf_(v);
      else if (MODE == M_SIG_PLAIN)   outF[(long)r * 64 + n] = sigmoidf_(v);
      else if (MODE == M_RELU128)     outB[(long)r * 128 + n] = f2bf(fmaxf(v, 0.f));
      else if (MODE == M_RESID)       outF[(long)r * 64 + n] += v;
      else                            outF[(long)r * 64 + n] += v * gate[(long)r * 64 + n];
    }
  }
}

// ---------------- trimul gated projection: out[c][rp] = (xn@W1+b1)*sigmoid(xn@W2+b2) ----------------
__global__ __launch_bounds__(256)
void gemm_gated_kernel(const bf16* __restrict__ A,
                       const bf16* __restrict__ Bpk1, const float* __restrict__ b1,
                       const bf16* __restrict__ Bpk2, const float* __restrict__ b2,
                       bf16* __restrict__ outT, int swap) {
  int lane = threadIdx.x & 31;
  int rt = blockIdx.x * 8 + (threadIdx.x >> 5);
  if (rt >= R2 / 16) return;
  int ml = lane & 15;
  int ka = (lane < 16) ? 0 : 16;
  f32x8 p[4], q[4];
#pragma unroll
  for (int t = 0; t < 4; ++t) { p[t] = zero8(); q[t] = zero8(); }
#pragma unroll
  for (int kb = 0; kb < 2; ++kb) {
    const bf16* ap = A + (long)(rt * 16 + ml) * 64 + kb * 32 + ka;
    bf16x16 af = ldA(ap);
#pragma unroll
    for (int nt = 0; nt < 4; ++nt) {
      p[nt] = wmma_bf16(af, ldB16(Bpk1 + (((long)kb * 4 + nt) * 32 + lane) * 16), p[nt]);
      q[nt] = wmma_bf16(af, ldB16(Bpk2 + (((long)kb * 4 + nt) * 32 + lane) * 16), q[nt]);
    }
  }
  int mrow = (lane < 16) ? 0 : 8;
#pragma unroll
  for (int nt = 0; nt < 4; ++nt) {
#pragma unroll
    for (int e = 0; e < 8; ++e) {
      int r = rt * 16 + mrow + e;
      int c = nt * 16 + ml;
      float val = (p[nt][e] + b1[c]) * sigmoidf_(q[nt][e] + b2[c]);
      int rp = swap ? ((r % NSEQ) * NSEQ + r / NSEQ) : r;
      outT[(long)c * R2 + rp] = f2bf(val);
    }
  }
}

// ---------------- fused triangle attention core ----------------
// One block per (fixed index, head). 384 threads = 12 waves; wave w owns S/O row-strip w.
// STARTING=1: per (i,h): S[j,k]=q[i,j,:].k[i,k,:], bias bb[h][k*192+j]
// STARTING=0: per (j,h): S[i,k]=q[i,j,:].k[k,j,:], bias bb[h][i*192+k]
template <int STARTING>
__global__ __launch_bounds__(384)
void attn_core_kernel(const bf16* __restrict__ qT, const bf16* __restrict__ kT,
                      const bf16* __restrict__ vT, const float* __restrict__ gbuf,
                      const float* __restrict__ bb, bf16* __restrict__ obuf) {
  extern __shared__ char smem[];
  bf16* VT = (bf16*)smem;                    // [32][208] V transposed (c-major)
  bf16* WL = (bf16*)(smem + 32 * 208 * 2);   // 12 waves x [16][192] softmaxed weights
  int h = blockIdx.x & 3;
  int fixed = blockIdx.x >> 2;
  int w = threadIdx.x >> 5, lane = threadIdx.x & 31;
  long hb = (long)h * R2 * 32;
  for (int t = threadIdx.x; t < NSEQ * 32; t += 384) {
    int kk = t >> 5, c = t & 31;
    long r = STARTING ? (long)fixed * NSEQ + kk : (long)kk * NSEQ + fixed;
    VT[c * 208 + kk] = vT[hb + r * 32 + c];
  }
  __syncthreads();
  int ml = lane & 15;
  int ka = (lane < 16) ? 0 : 16;
  int kc = (lane < 16) ? 0 : 16;
  int mrow = (lane < 16) ? 0 : 8;
  // S strip: 12 k-tiles, single K=32 (CH) WMMA each
  long rq = STARTING ? (long)fixed * NSEQ + (w * 16 + ml) : (long)(w * 16 + ml) * NSEQ + fixed;
  bf16x16 af = ldA(qT + hb + rq * 32 + ka);
  f32x8 s[12];
#pragma unroll
  for (int kt = 0; kt < 12; ++kt) {
    long rk = STARTING ? (long)fixed * NSEQ + (kt * 16 + ml) : (long)(kt * 16 + ml) * NSEQ + fixed;
    s[kt] = wmma_bf16(af, ldB16(kT + hb + rk * 32 + kc), zero8());
  }
  const float scale = 0.17677669529663687f;  // 1/sqrt(32)
#pragma unroll
  for (int kt = 0; kt < 12; ++kt) {
#pragma unroll
    for (int e = 0; e < 8; ++e) {
      int jr = w * 16 + mrow + e;
      int kg = kt * 16 + ml;
      float bias = STARTING ? bb[(long)h * R2 + (long)kg * NSEQ + jr]
                            : bb[(long)h * R2 + (long)jr * NSEQ + kg];
      s[kt][e] = s[kt][e] * scale + bias;
    }
  }
  // softmax over k (12 tiles x 16 lanes within each half-wave group)
#pragma unroll
  for (int e = 0; e < 8; ++e) {
    float m = s[0][e];
#pragma unroll
    for (int kt = 1; kt < 12; ++kt) m = fmaxf(m, s[kt][e]);
#pragma unroll
    for (int xm = 1; xm <= 8; xm <<= 1) m = fmaxf(m, __shfl_xor(m, xm, 32));
    float t = 0.f;
#pragma unroll
    for (int kt = 0; kt < 12; ++kt) { float ev = __expf(s[kt][e] - m); s[kt][e] = ev; t += ev; }
#pragma unroll
    for (int xm = 1; xm <= 8; xm <<= 1) t += __shfl_xor(t, xm, 32);
    float inv = 1.f / t;
#pragma unroll
    for (int kt = 0; kt < 12; ++kt) s[kt][e] *= inv;
  }
  // spill W strip to LDS in row-major for A-fragment reloads
  bf16* wl = WL + (long)w * 16 * NSEQ;
#pragma unroll
  for (int kt = 0; kt < 12; ++kt)
#pragma unroll
    for (int e = 0; e < 8; ++e)
      wl[(mrow + e) * NSEQ + kt * 16 + ml] = f2bf(s[kt][e]);
  __syncthreads();
  // O = W @ V, gate, store head-major
#pragma unroll
  for (int ct = 0; ct < 2; ++ct) {
    f32x8 oa = zero8();
#pragma unroll
    for (int kb = 0; kb < 6; ++kb) {
      bf16x16 aw = ldA(wl + ml * NSEQ + kb * 32 + ka);
      bf16x16 bv = ldB16(VT + (ct * 16 + ml) * 208 + kb * 32 + kc);
      oa = wmma_bf16(aw, bv, oa);
    }
#pragma unroll
    for (int e = 0; e < 8; ++e) {
      int jr = w * 16 + mrow + e;
      int c = ct * 16 + ml;
      long r = STARTING ? (long)fixed * NSEQ + jr : (long)jr * NSEQ + fixed;
      float g = gbuf[hb + r * 32 + c];
      obuf[hb + r * 32 + c] = f2bf(oa[e] * g);
    }
  }
}

// ---------------- trimul einsum: per channel c, O_c = A_c @ B_c^T, 192x192x192 ----------------
__global__ __launch_bounds__(256)
void trimul_mm_kernel(const bf16* __restrict__ aT, const bf16* __restrict__ bT,
                      float* __restrict__ outO) {
  int lane = threadIdx.x & 31;
  int wblk = threadIdx.x >> 5;
  int c = blockIdx.x / 18, grp = blockIdx.x % 18;
  int wt = grp * 8 + wblk;                 // 0..143 tiles of 16x16
  int rt = wt / 12, nt = wt % 12;
  const bf16* A = aT + (long)c * R2;
  const bf16* B = bT + (long)c * R2;
  int ml = lane & 15, ka = (lane < 16) ? 0 : 16, kc = (lane < 16) ? 0 : 16, mrow = (lane < 16) ? 0 : 8;
  f32x8 acc = zero8();
#pragma unroll
  for (int kb = 0; kb < 6; ++kb) {
    bf16x16 af = ldA(A + (long)(rt * 16 + ml) * NSEQ + kb * 32 + ka);
    bf16x16 bf_ = ldB16(B + (long)(nt * 16 + ml) * NSEQ + kb * 32 + kc);
    acc = wmma_bf16(af, bf_, acc);
  }
#pragma unroll
  for (int e = 0; e < 8; ++e) {
    int i = rt * 16 + mrow + e, j = nt * 16 + ml;
    outO[((long)i * NSEQ + j) * 64 + c] = acc[e];
  }
}

// ================= host side =================
static constexpr size_t MB_ = 1u << 20;
static constexpr size_t OF_XN = 0;             // bf16 [36864,64]         4.72MB
static constexpr size_t OF_PK = 5 * MB_;       // packed weights          <1MB
// attention stage scratch
static constexpr size_t S_QT = 6 * MB_,  S_KT = 16 * MB_, S_VT = 26 * MB_;
static constexpr size_t S_G  = 36 * MB_, S_BB = 55 * MB_, S_OB = 56 * MB_;
// trimul stage scratch (reuses attention region)
static constexpr size_t S_AT = 6 * MB_,  S_BT = 11 * MB_, S_TO = 16 * MB_;
static constexpr size_t S_ON = 26 * MB_, S_TG = 31 * MB_;
// transition
static constexpr size_t S_H1 = 6 * MB_;

extern "C" void kernel_launch(void* const* d_in, const int* in_sizes, int n_in,
                              void* d_out, int out_size, void* d_ws, size_t ws_size,
                              hipStream_t stream) {
  (void)in_sizes; (void)n_in; (void)out_size; (void)ws_size;
  // Input order (setup_inputs dict insertion order):
  // 0:x2d | p_as 1:ns 2:nb 3:wq 4:wk 5:wv 6:wb 7:wg 8:bg 9:wo 10:bo | p_ae 11..20
  // p_mo 21:n1s 22:n1b 23:wi 24:bi 25:wj 26:bj 27:wis 28:bis 29:wjs 30:bjs 31:n2s 32:n2b 33:wp 34:bp 35:wg 36:bg
  // p_mi 37..52 | p_tr 53:ns 54:nb 55:w1 56:b1 57:w2 58:b2
  auto F = [&](int i) { return (const float*)d_in[i]; };
  float* x = (float*)d_out;
  char* ws = (char*)d_ws;
  hipMemcpyAsync(x, d_in[0], (size_t)R2 * 64 * 4, hipMemcpyDeviceToDevice, stream);

  auto pack = [&](const float* W, size_t off, int K, int Nn, int perm) {
    int kblks = (K + 31) / 32, ntc = (Nn + 15) / 16;
    int total = kblks * ntc * 512;
    pack_kernel<<<(total + 255) / 256, 256, 0, stream>>>(W, (bf16*)(ws + off), K, Nn, ntc, perm, total);
  };

  // ---- pack all weights ----
  for (int L = 0; L < 2; ++L) {
    int b = 1 + L * 10;
    size_t pb = OF_PK + (size_t)L * 98304;
    pack(F(b + 2), pb + 0,     64, 128, 1);  // wq (head-major cols)
    pack(F(b + 3), pb + 16384, 64, 128, 1);  // wk
    pack(F(b + 4), pb + 32768, 64, 128, 1);  // wv
    pack(F(b + 6), pb + 49152, 64, 128, 1);  // wg
    pack(F(b + 8), pb + 65536, 128, 64, 2);  // wo (head-major rows)
    pack(F(b + 5), pb + 81920, 64, 4, 0);    // wb (padded to N=16)
  }
  for (int L = 0; L < 2; ++L) {
    int b = 21 + L * 16;
    size_t pb = OF_PK + 196608 + (size_t)L * 65536;
    pack(F(b + 2),  pb + 0,     64, 64, 0);  // wi
    pack(F(b + 6),  pb + 8192,  64, 64, 0);  // wis
    pack(F(b + 4),  pb + 16384, 64, 64, 0);  // wj
    pack(F(b + 8),  pb + 24576, 64, 64, 0);  // wjs
    pack(F(b + 12), pb + 32768, 64, 64, 0);  // wp
    pack(F(b + 14), pb + 40960, 64, 64, 0);  // wg
  }
  size_t ptr_ = OF_PK + 327680;
  pack(F(55), ptr_ + 0,     64, 128, 0);     // w1
  pack(F(57), ptr_ + 16384, 128, 64, 0);     // w2

  bf16* xn = (bf16*)(ws + OF_XN);
  const int GB = 288;  // 2304 waves = 36864/16 row tiles

  // ---- two triangle attentions ----
  for (int L = 0; L < 2; ++L) {
    int b = 1 + L * 10;
    size_t pb = OF_PK + (size_t)L * 98304;
    bf16* qT = (bf16*)(ws + S_QT); bf16* kT = (bf16*)(ws + S_KT); bf16* vT = (bf16*)(ws + S_VT);
    float* gB = (float*)(ws + S_G); float* bbF = (float*)(ws + S_BB); bf16* oB = (bf16*)(ws + S_OB);
    ln64_kernel<<<4608, 256, 0, stream>>>(x, F(b), F(b + 1), xn);
    gemm_kernel<2, 8, M_BF16_SPLITH, 0><<<GB, 256, 0, stream>>>(
        xn, 64, 32, (bf16*)(ws + pb + 0), nullptr, nullptr, qT, nullptr);
    gemm_kernel<2, 8, M_BF16_SPLITH, 0><<<GB, 256, 0, stream>>>(
        xn, 64, 32, (bf16*)(ws + pb + 16384), nullptr, nullptr, kT, nullptr);
    gemm_kernel<2, 8, M_BF16_SPLITH, 0><<<GB, 256, 0, stream>>>(
        xn, 64, 32, (bf16*)(ws + pb + 32768), nullptr, nullptr, vT, nullptr);
    gemm_kernel<2, 8, M_SIG_SPLITH, 2><<<GB, 256, 0, stream>>>(
        xn, 64, 32, (bf16*)(ws + pb + 49152), F(b + 7), gB, nullptr, nullptr);
    gemm_kernel<2, 1, M_F32T4, 0><<<GB, 256, 0, stream>>>(
        xn, 64, 32, (bf16*)(ws + pb + 81920), nullptr, bbF, nullptr, nullptr);
    size_t lds = 32 * 208 * 2 + 12 * 16 * NSEQ * 2;  // 87040 B
    if (L == 0) attn_core_kernel<1><<<768, 384, lds, stream>>>(qT, kT, vT, gB, bbF, oB);
    else        attn_core_kernel<0><<<768, 384, lds, stream>>>(qT, kT, vT, gB, bbF, oB);
    gemm_kernel<4, 4, M_RESID, 1><<<GB, 256, 0, stream>>>(
        oB, 32, (long)R2 * 32, (bf16*)(ws + pb + 65536), F(b + 9), x, nullptr, nullptr);
  }

  // ---- two triangle multiplications (L=0 outgoing, L=1 incoming) ----
  for (int L = 0; L < 2; ++L) {
    int b = 21 + L * 16;
    size_t pb = OF_PK + 196608 + (size_t)L * 65536;
    int swap = L;
    bf16* aT = (bf16*)(ws + S_AT); bf16* bT = (bf16*)(ws + S_BT);
    float* toF = (float*)(ws + S_TO); bf16* on = (bf16*)(ws + S_ON); float* tg = (float*)(ws + S_TG);
    ln64_kernel<<<4608, 256, 0, stream>>>(x, F(b), F(b + 1), xn);
    gemm_gated_kernel<<<GB, 256, 0, stream>>>(xn, (bf16*)(ws + pb + 0),     F(b + 3),
                                              (bf16*)(ws + pb + 8192),  F(b + 7), aT, swap);
    gemm_gated_kernel<<<GB, 256, 0, stream>>>(xn, (bf16*)(ws + pb + 16384), F(b + 5),
                                              (bf16*)(ws + pb + 24576), F(b + 9), bT, swap);
    gemm_kernel<2, 4, M_SIG_PLAIN, 1><<<GB, 256, 0, stream>>>(
        xn, 64, 32, (bf16*)(ws + pb + 40960), F(b + 15), tg, nullptr, nullptr);
    trimul_mm_kernel<<<64 * 18, 256, 0, stream>>>(aT, bT, toF);
    ln64_kernel<<<4608, 256, 0, stream>>>(toF, F(b + 10), F(b + 11), on);
    gemm_kernel<2, 4, M_RESID_GATE, 1><<<GB, 256, 0, stream>>>(
        on, 64, 32, (bf16*)(ws + pb + 32768), F(b + 13), x, nullptr, tg);
  }

  // ---- transition ----
  {
    bf16* h1 = (bf16*)(ws + S_H1);
    ln64_kernel<<<4608, 256, 0, stream>>>(x, F(53), F(54), xn);
    gemm_kernel<2, 8, M_RELU128, 1><<<GB, 256, 0, stream>>>(
        xn, 64, 32, (bf16*)(ws + ptr_ + 0), F(56), nullptr, h1, nullptr);
    gemm_kernel<4, 4, M_RESID, 1><<<GB, 256, 0, stream>>>(
        h1, 128, 32, (bf16*)(ws + ptr_ + 16384), F(58), x, nullptr, nullptr);
  }
}
